// GPT_69286412419265
// MI455X (gfx1250) — compile-verified, matrix operand order checked by reading, not run on hardware
//
#include <hip/hip_runtime.h>
#include <math.h>

// ---------------- model constants (match reference) ----------------
#define KDIM   512
#define HEADS  8
#define NBLK   4
#define VOCABN 32000
#define BATCHN 4
#define TSEQ   512
#define KH     (KDIM * HEADS)     // 4096
#define MROWS  (BATCHN * TSEQ)    // 2048
#define FFD    (4 * KDIM)         // 2048

typedef __bf16 bf16_t;
typedef bf16_t v16bf __attribute__((ext_vector_type(16)));
typedef float  v8f   __attribute__((ext_vector_type(8)));

__device__ __forceinline__ unsigned short f32_to_bf16(float f) {
  unsigned int u = __float_as_uint(f);
  u += 0x7fffu + ((u >> 16) & 1u);           // round-to-nearest-even
  return (unsigned short)(u >> 16);
}

union Frag { uint4 q[2]; v16bf v; };

// =====================================================================
// WMMA GEMM, all operands bf16 & k-contiguous:
//   C[z][m][n] = sum_k A[z][m][k] * B[z][n][k]   (B given transposed, [N,K])
// Wave tile = 32(M) x 64(N): 2 M-tiles x 4 N-tiles, 8 f32 accumulators.
// Block = 8 waves -> 256(M) x 64(N) macro-tile. Per K-step(32):
// 12x global_load_b128 feed 8x v_wmma_f32_16x16x32_bf16.
//   z decomposes zb = z/Hdiv, zh = z%Hdiv (batch, head) slice offsets.
//   epi: 0 = f32 store (+bias), 1 = bf16 store, 3 = exact-GELU->bf16,
//        4 = bf16 scatter into V-transposed layout [b,h,d,s].
// All M are multiples of 256, N of 64, K of 32 -> no predication (EXEC all-1s).
// =====================================================================
__global__ __launch_bounds__(256) void gemm_wmma_bf16(
    const unsigned short* __restrict__ A, const unsigned short* __restrict__ B,
    const float* __restrict__ bias, float* __restrict__ Cf,
    unsigned short* __restrict__ Cb,
    int M, int N, int Kd, int Hdiv,
    long lda, long a_sb, long a_sh,
    long ldb, long b_sb, long b_sh,
    long csm, long c_sb, long c_sh, int epi) {
  const int lane = threadIdx.x & 31;
  const int wave = threadIdx.x >> 5;
  const int r    = lane & 15;          // row (A) / col (B,C) within tile
  const int half = lane >> 4;          // K-half selector per ISA layout
  const int m0 = blockIdx.y * 256 + wave * 32;
  const int n0 = blockIdx.x * 64;
  const int zb = (int)blockIdx.z / Hdiv, zh = (int)blockIdx.z % Hdiv;
  const size_t abase = (size_t)zb * a_sb + (size_t)zh * a_sh;
  const size_t bbase = (size_t)zb * b_sb + (size_t)zh * b_sh;
  const size_t cbase = (size_t)zb * c_sb + (size_t)zh * c_sh;
  if (m0 >= M || n0 >= N) return;

  const v8f zero = {0.f, 0.f, 0.f, 0.f, 0.f, 0.f, 0.f, 0.f};
  v8f acc[2][4] = {{zero, zero, zero, zero}, {zero, zero, zero, zero}};

  // A fragment rows: lane r -> M = m0 + r (+16); K halves split by lane>>4.
  const unsigned short* ar0 = A + abase + (size_t)(m0 + r) * lda + half * 8;
  const unsigned short* ar1 = ar0 + (size_t)16 * lda;
  // B fragment cols: lane r -> N = n0 + r; same K-half split.
  const unsigned short* br  = B + bbase + (size_t)(n0 + r) * ldb + half * 8;

  for (int k0 = 0; k0 < Kd; k0 += 32) {
    Frag a0, a1;
    a0.q[0] = *(const uint4*)(ar0 + k0);
    a0.q[1] = *(const uint4*)(ar0 + k0 + 16);
    a1.q[0] = *(const uint4*)(ar1 + k0);
    a1.q[1] = *(const uint4*)(ar1 + k0 + 16);
#pragma unroll
    for (int j = 0; j < 4; ++j) {
      Frag b;
      const unsigned short* bp = br + (size_t)j * 16 * ldb + k0;
      b.q[0] = *(const uint4*)(bp);
      b.q[1] = *(const uint4*)(bp + 16);
      acc[0][j] = __builtin_amdgcn_wmma_f32_16x16x32_bf16(
          false, a0.v, false, b.v, (short)0, acc[0][j], false, false);
      acc[1][j] = __builtin_amdgcn_wmma_f32_16x16x32_bf16(
          false, a1.v, false, b.v, (short)0, acc[1][j], false, false);
    }
  }

  // Epilogue. C/D layout: VGPR v, lanes 0-15 -> M = base+v, lanes 16-31 -> base+8+v.
#pragma unroll
  for (int mi = 0; mi < 2; ++mi) {
#pragma unroll
    for (int j = 0; j < 4; ++j) {
      const int n = n0 + j * 16 + r;
      const float bn = bias ? bias[n] : 0.0f;
#pragma unroll
      for (int vv = 0; vv < 8; ++vv) {
        const int m = m0 + mi * 16 + half * 8 + vv;
        float val = acc[mi][j][vv] + bn;
        if (epi == 3)  // exact GELU: 0.5*x*(1+erf(x/sqrt(2)))
          val = 0.5f * val * (1.0f + erff(val * 0.7071067811865476f));
        if (epi == 4) {  // scatter V: [b,t,h,d] tile -> vt[b,h,d,s]
          const int bb = m / TSEQ, ss = m % TSEQ;
          const int hh = n / KDIM, dd = n % KDIM;
          Cb[(((size_t)bb * HEADS + hh) * KDIM + dd) * TSEQ + ss] = f32_to_bf16(val);
        } else {
          const size_t coff = cbase + (size_t)m * csm + n;
          if (epi == 0) Cf[coff] = val;
          else          Cb[coff] = f32_to_bf16(val);
        }
      }
    }
  }
}

// ---------------- f32 [K,N] row-major -> bf16 [N,K] (LDS-tiled transpose) ----------------
__global__ __launch_bounds__(256) void k_transcvt(
    const float* __restrict__ W, unsigned short* __restrict__ Wt, int Kd, int N) {
  __shared__ float tile[32][33];
  const int kt = blockIdx.y * 32, nt = blockIdx.x * 32;
  const int tx = threadIdx.x & 31, ty = threadIdx.x >> 5;  // 32 x 8
#pragma unroll
  for (int yy = 0; yy < 32; yy += 8)
    tile[ty + yy][tx] = W[(size_t)(kt + ty + yy) * N + nt + tx];
  __syncthreads();
#pragma unroll
  for (int yy = 0; yy < 32; yy += 8)
    Wt[(size_t)(nt + ty + yy) * Kd + kt + tx] = f32_to_bf16(tile[tx][ty + yy]);
}

// ---------------- embedding + sinusoidal positional encoding ----------------
__global__ __launch_bounds__(256) void k_embed(
    const int* __restrict__ x, const float* __restrict__ embW,
    float* __restrict__ h, unsigned short* __restrict__ hb) {
  const int idx = blockIdx.x * blockDim.x + threadIdx.x;  // over B*T*K
  if (idx >= MROWS * KDIM) return;
  const int c  = idx % KDIM;
  const int bt = idx / KDIM;
  const int t  = bt % TSEQ;
  const int tok = x[bt];
  const int i = c >> 1;  // div = 10000 ** (4*i/K)
  const float ang = (float)t * powf(10000.0f, -4.0f * (float)i / (float)KDIM);
  const float pe = (c & 1) ? cosf(ang) : sinf(ang);
  const float v = embW[(size_t)tok * KDIM + c] + pe;
  h[idx]  = v;
  hb[idx] = f32_to_bf16(v);
}

// ---------------- causal softmax over score rows; writes bf16 probs ----------------
__global__ __launch_bounds__(256) void k_softmax(
    const float* __restrict__ att, unsigned short* __restrict__ attb) {
  const int row = blockIdx.x;          // ((b*H + h) * T + q)
  const int q   = row % TSEQ;
  const float scale = 0.044194173824159216f;  // 1/sqrt(512)
  const float* p = att + (size_t)row * TSEQ;
  unsigned short* o = attb + (size_t)row * TSEQ;
  __shared__ float red[256];
  const int tid = threadIdx.x;

  float mx = -INFINITY;
  for (int s = tid; s < TSEQ; s += 256)
    if (s <= q) mx = fmaxf(mx, p[s] * scale);
  red[tid] = mx; __syncthreads();
  for (int off = 128; off; off >>= 1) {
    if (tid < off) red[tid] = fmaxf(red[tid], red[tid + off]);
    __syncthreads();
  }
  mx = red[0]; __syncthreads();

  float sum = 0.f;
  for (int s = tid; s < TSEQ; s += 256)
    if (s <= q) sum += expf(p[s] * scale - mx);
  red[tid] = sum; __syncthreads();
  for (int off = 128; off; off >>= 1) {
    if (tid < off) red[tid] += red[tid + off];
    __syncthreads();
  }
  const float inv = 1.0f / red[0];

  for (int s = tid; s < TSEQ; s += 256) {
    const float e = (s <= q) ? expf(p[s] * scale - mx) * inv : 0.0f;
    o[s] = f32_to_bf16(e);
  }
}

// ---------------- (optional residual add) + LayerNorm; emits f32 + bf16 ----------------
__global__ __launch_bounds__(256) void k_resln(
    const float* __restrict__ xin, const float* __restrict__ resid,
    const float* __restrict__ w, const float* __restrict__ b,
    float* __restrict__ hout, unsigned short* __restrict__ hbout) {
  const int row = blockIdx.x;
  const int tid = threadIdx.x;
  const size_t base = (size_t)row * KDIM;
  __shared__ float red[256];

  float v0 = xin[base + tid];
  float v1 = xin[base + tid + 256];
  if (resid) { v0 += resid[base + tid]; v1 += resid[base + tid + 256]; }

  red[tid] = v0 + v1; __syncthreads();
  for (int off = 128; off; off >>= 1) {
    if (tid < off) red[tid] += red[tid + off];
    __syncthreads();
  }
  const float mean = red[0] * (1.0f / KDIM);
  __syncthreads();

  const float d0 = v0 - mean, d1 = v1 - mean;
  red[tid] = d0 * d0 + d1 * d1; __syncthreads();
  for (int off = 128; off; off >>= 1) {
    if (tid < off) red[tid] += red[tid + off];
    __syncthreads();
  }
  const float inv = rsqrtf(red[0] * (1.0f / KDIM) + 1e-5f);

  const float o0 = d0 * inv * w[tid]       + b[tid];
  const float o1 = d1 * inv * w[tid + 256] + b[tid + 256];
  hout[base + tid]        = o0;
  hout[base + tid + 256]  = o1;
  hbout[base + tid]       = f32_to_bf16(o0);
  hbout[base + tid + 256] = f32_to_bf16(o1);
}

// =====================================================================
extern "C" void kernel_launch(void* const* d_in, const int* in_sizes, int n_in,
                              void* d_out, int out_size, void* d_ws, size_t ws_size,
                              hipStream_t stream) {
  (void)in_sizes; (void)n_in; (void)out_size; (void)ws_size;
  const int*   x    = (const int*)d_in[0];
  const float* embW = (const float*)d_in[1];
  const float* Wq   = (const float*)d_in[2];
  const float* Wk   = (const float*)d_in[3];
  const float* Wv   = (const float*)d_in[4];
  const float* Wu   = (const float*)d_in[5];
  const float* bu   = (const float*)d_in[6];
  const float* W1   = (const float*)d_in[7];
  const float* b1   = (const float*)d_in[8];
  const float* W2   = (const float*)d_in[9];
  const float* b2   = (const float*)d_in[10];
  const float* ln1w = (const float*)d_in[11];
  const float* ln1b = (const float*)d_in[12];
  const float* ln2w = (const float*)d_in[13];
  const float* ln2b = (const float*)d_in[14];
  const float* lnfw = (const float*)d_in[15];
  const float* lnfb = (const float*)d_in[16];
  const float* UeW  = (const float*)d_in[17];
  const float* Ueb  = (const float*)d_in[18];
  float* out = (float*)d_out;

  // ---- workspace carve-out (~162 MB total) ----
  char* ws = (char*)d_ws;
  size_t off = 0;
  auto take = [&](size_t bytes) -> char* {
    char* p = ws + off;
    off += (bytes + 255) & ~(size_t)255;
    return p;
  };
  unsigned short* hb   = (unsigned short*)take((size_t)MROWS * KDIM * 2);          //  2 MB
  unsigned short* qb   = (unsigned short*)take((size_t)MROWS * KH * 2);            // 16 MB
  unsigned short* kb   = (unsigned short*)take((size_t)MROWS * KH * 2);            // 16 MB
  unsigned short* vt   = (unsigned short*)take((size_t)MROWS * KH * 2);            // 16 MB [b,h,d,s]
  float*          att  = (float*)take((size_t)BATCHN * HEADS * TSEQ * TSEQ * 4);   // 32 MB
  unsigned short* attb = (unsigned short*)take((size_t)BATCHN * HEADS * TSEQ * TSEQ * 2); // 16 MB
  unsigned short* yb   = (unsigned short*)take((size_t)MROWS * KH * 2);            // 16 MB
  float*          u    = (float*)take((size_t)MROWS * KDIM * 4);                   //  4 MB
  unsigned short* g    = (unsigned short*)take((size_t)MROWS * FFD * 2);           //  8 MB
  float*          h    = (float*)take((size_t)MROWS * KDIM * 4);                   //  4 MB
  unsigned short* wt   = (unsigned short*)take((size_t)VOCABN * KDIM * 2);         // 32 MB shared W^T

  const dim3 blk(256);
  auto gemm = [&](const unsigned short* A, const unsigned short* B,
                  const float* bias, float* Cf, unsigned short* Cb, int M, int N,
                  int Kd, int Hdiv, long lda, long a_sb, long a_sh, long ldb,
                  long b_sb, long b_sh, long csm, long c_sb, long c_sh, int Z,
                  int epi) {
    dim3 grid(N / 64, M / 256, Z);
    gemm_wmma_bf16<<<grid, blk, 0, stream>>>(A, B, bias, Cf, Cb, M, N, Kd, Hdiv,
                                             lda, a_sb, a_sh, ldb, b_sb, b_sh,
                                             csm, c_sb, c_sh, epi);
  };
  auto tc = [&](const float* W, int Kd, int N) {  // -> wt
    dim3 grid(N / 32, Kd / 32);
    k_transcvt<<<grid, blk, 0, stream>>>(W, wt, Kd, N);
  };

  // ---- embedding + positional encoding ----
  k_embed<<<(MROWS * KDIM + 255) / 256, 256, 0, stream>>>(x, embW, h, hb);

  for (int i = 0; i < NBLK; ++i) {
    // Q/K/V projections: [2048,512] @ [512,4096] -> bf16 (V scattered to [b,h,d,s])
    tc(Wq + (size_t)i * KDIM * KH, KDIM, KH);
    gemm(hb, wt, nullptr, nullptr, qb, MROWS, KH, KDIM, 1,
         KDIM, 0, 0, KDIM, 0, 0, KH, 0, 0, 1, 1);
    tc(Wk + (size_t)i * KDIM * KH, KDIM, KH);
    gemm(hb, wt, nullptr, nullptr, kb, MROWS, KH, KDIM, 1,
         KDIM, 0, 0, KDIM, 0, 0, KH, 0, 0, 1, 1);
    tc(Wv + (size_t)i * KDIM * KH, KDIM, KH);
    gemm(hb, wt, nullptr, nullptr, vt, MROWS, KH, KDIM, 1,
         KDIM, 0, 0, KDIM, 0, 0, 0, 0, 0, 1, 4);

    // scores[b,h,q,s] = sum_d q[b,q,h,d] * k[b,s,h,d]   (per b,h slice)
    gemm(qb, kb, nullptr, att, nullptr, TSEQ, TSEQ, KDIM, HEADS,
         KH, (long)TSEQ * KH, KDIM,
         KH, (long)TSEQ * KH, KDIM,
         TSEQ, (long)HEADS * TSEQ * TSEQ, (long)TSEQ * TSEQ,
         BATCHN * HEADS, 0);

    k_softmax<<<BATCHN * HEADS * TSEQ, 256, 0, stream>>>(att, attb);

    // y[b,q,h,d] = sum_s att[b,h,q,s] * vt[b,h,d,s]
    gemm(attb, vt, nullptr, nullptr, yb, TSEQ, KDIM, TSEQ, HEADS,
         TSEQ, (long)HEADS * TSEQ * TSEQ, (long)TSEQ * TSEQ,
         TSEQ, (long)HEADS * KDIM * TSEQ, (long)KDIM * TSEQ,
         KH, (long)TSEQ * KH, KDIM,
         BATCHN * HEADS, 1);

    // u = y @ Wu + bu : [2048,4096] @ [4096,512]
    tc(Wu + (size_t)i * KH * KDIM, KH, KDIM);
    gemm(yb, wt, bu + (size_t)i * KDIM, u, nullptr, MROWS, KDIM, KH, 1,
         KH, 0, 0, KH, 0, 0, KDIM, 0, 0, 1, 0);
    k_resln<<<MROWS, 256, 0, stream>>>(u, h, ln1w + (size_t)i * KDIM,
                                       ln1b + (size_t)i * KDIM, h, hb);

    // FFN: gelu(h @ W1 + b1) @ W2 + b2
    tc(W1 + (size_t)i * KDIM * FFD, KDIM, FFD);
    gemm(hb, wt, b1 + (size_t)i * FFD, nullptr, g, MROWS, FFD, KDIM, 1,
         KDIM, 0, 0, KDIM, 0, 0, FFD, 0, 0, 1, 3);
    tc(W2 + (size_t)i * FFD * KDIM, FFD, KDIM);
    gemm(g, wt, b2 + (size_t)i * KDIM, u, nullptr, MROWS, KDIM, FFD, 1,
         FFD, 0, 0, FFD, 0, 0, KDIM, 0, 0, 1, 0);
    k_resln<<<MROWS, 256, 0, stream>>>(u, h, ln2w + (size_t)i * KDIM,
                                       ln2b + (size_t)i * KDIM, h, hb);
  }

  // final LayerNorm, then unembed: [2048,512] @ [512,32000] + b -> d_out
  k_resln<<<MROWS, 256, 0, stream>>>(h, nullptr, lnfw, lnfb, h, hb);
  tc(UeW, KDIM, VOCABN);
  gemm(hb, wt, Ueb, out, nullptr, MROWS, VOCABN, KDIM, 1,
       KDIM, 0, 0, KDIM, 0, 0, VOCABN, 0, 0, 1, 0);
}
